// GraphSAGE_34557306863778
// MI455X (gfx1250) — matrix operand order repaired and checked by reading, not compiled, not run on hardware
//
#include <hip/hip_runtime.h>

typedef __bf16 bf16_t;
typedef __attribute__((ext_vector_type(16))) __bf16 v16bf;
typedef __attribute__((ext_vector_type(8)))  __bf16 v8bf;
typedef __attribute__((ext_vector_type(8)))  float  v8f;

#define NNODES 50000
#define SS     10
#define DOUT   256

// ---------------------------------------------------------------------------
// Convert W [K, Nc] f32 (row-major) -> Wt [Nc, K] bf16 (transposed) so that a
// WMMA B-fragment for a given output column reads a contiguous K-run.
// ---------------------------------------------------------------------------
__global__ void wt_convert_kernel(const float* __restrict__ W,
                                  bf16_t* __restrict__ Wt,
                                  int K, int Nc) {
  int i = blockIdx.x * blockDim.x + threadIdx.x;
  if (i >= K * Nc) return;
  int k = i / Nc;
  int c = i - k * Nc;
  Wt[(size_t)c * K + k] = (bf16_t)W[(size_t)k * Nc + c];
}

// ---------------------------------------------------------------------------
// One GraphSAGE layer: masked-mean aggregate (f32) -> concat row (bf16, LDS)
// -> [16 x 2*DIN] @ Wt^T via v_wmma_f32_16x16x32_bf16 -> bias + ReLU -> out.
// Block = 16 nodes, 256 threads = 8 waves; wave w computes column tiles
// {2w, 2w+1} of the 256 output columns.
// ---------------------------------------------------------------------------
template <int DIN>
__global__ __launch_bounds__(256) void sage_layer_kernel(
    const float* __restrict__ feats,   // [N, DIN]
    const int* __restrict__ nidx,      // [N, S]
    const float* __restrict__ nmask,   // [N, S]
    const bf16_t* __restrict__ Wt,     // [DOUT, 2*DIN] (transposed bf16 W)
    const float* __restrict__ bias,    // [DOUT]
    float* __restrict__ out)           // [N, DOUT]
{
  constexpr int KDIM = 2 * DIN;
  constexpr int PAD  = 8;             // bf16 pad to stagger LDS banks
  constexpr int ROW  = KDIM + PAD;
  __shared__ bf16_t combA[16 * ROW];  // [16 nodes][KDIM] concat(self, mean)

  const int tid  = threadIdx.x;
  const int g    = tid >> 4;          // node slot within tile (0..15)
  const int t    = tid & 15;          // lane within node group
  const int node = blockIdx.x * 16 + g;

  // ---- Phase A: aggregation (full f32), write bf16 rows to LDS ----
  float msum = 0.f;
#pragma unroll
  for (int s = 0; s < SS; ++s) msum += nmask[node * SS + s];
  const float inv = 1.0f / msum;      // reference guarantees msum >= 1

  const float4* feats4 = (const float4*)feats;
  constexpr int PASSES = DIN / 64;    // 16 threads x float4 = 64 floats/pass
#pragma unroll
  for (int p = 0; p < PASSES; ++p) {
    const int d4 = p * 16 + t;
    // self features -> comb[g][0..DIN)
    float4 sv = feats4[(size_t)node * (DIN / 4) + d4];
    bf16_t* dstS = &combA[g * ROW + d4 * 4];
    dstS[0] = (bf16_t)sv.x; dstS[1] = (bf16_t)sv.y;
    dstS[2] = (bf16_t)sv.z; dstS[3] = (bf16_t)sv.w;
    // masked-mean of neighbors -> comb[g][DIN..2*DIN)
    float ax = 0.f, ay = 0.f, az = 0.f, aw = 0.f;
#pragma unroll
    for (int s = 0; s < SS; ++s) {
      const float w  = nmask[node * SS + s];
      const int   nb = nidx[node * SS + s];
      float4 nv = feats4[(size_t)nb * (DIN / 4) + d4];
      ax = fmaf(w, nv.x, ax); ay = fmaf(w, nv.y, ay);
      az = fmaf(w, nv.z, az); aw = fmaf(w, nv.w, aw);
    }
    bf16_t* dstN = &combA[g * ROW + DIN + d4 * 4];
    dstN[0] = (bf16_t)(ax * inv); dstN[1] = (bf16_t)(ay * inv);
    dstN[2] = (bf16_t)(az * inv); dstN[3] = (bf16_t)(aw * inv);
  }
  __syncthreads();

  // ---- Phase B: WMMA GEMM, 16x16 output tiles ----
  const int wave  = tid >> 5;         // 0..7
  const int lane  = tid & 31;
  const int lm    = lane & 15;        // M (A) / N (B) index
  const int khalf = lane >> 4;        // lane-half selects K sub-block

#pragma unroll
  for (int ct2 = 0; ct2 < 2; ++ct2) {
    const int c0 = (wave * 2 + ct2) * 16;
    const float bcol = bias[c0 + lm];
    v8f acc;
#pragma unroll
    for (int i = 0; i < 8; ++i) acc[i] = bcol;

    const bf16_t* arow = &combA[lm * ROW];
    const bf16_t* wrow = Wt + (size_t)(c0 + lm) * KDIM;

#pragma unroll 4
    for (int kb = 0; kb < KDIM; kb += 32) {
      // A fragment (contiguous k order, same as B): lanes<16 hold K {0..15},
      // lanes>=16 hold K {16..31} -> two contiguous 16B LDS loads.
      v8bf alo = *(const v8bf*)(arow + kb + khalf * 16);
      v8bf ahi = *(const v8bf*)(arow + kb + khalf * 16 + 8);
      v16bf afrag;
#pragma unroll
      for (int i = 0; i < 8; ++i) { afrag[i] = alo[i]; afrag[i + 8] = ahi[i]; }
      // B fragment (32x16): lane = N, halves sweep 16 contiguous K values
      // per lane-half -> one 32B contiguous read from the transposed weights.
      v16bf bfrag = *(const v16bf*)(wrow + kb + khalf * 16);

      acc = __builtin_amdgcn_wmma_f32_16x16x32_bf16(
          false, afrag, false, bfrag, (short)0, acc, false, false);
    }

    // C/D layout: VGPR i, lanes<16 -> M=i, lanes>=16 -> M=i+8; N = lane&15.
#pragma unroll
    for (int i = 0; i < 8; ++i) {
      const int row = blockIdx.x * 16 + i + khalf * 8;
      float v = acc[i];
      out[(size_t)row * DOUT + c0 + lm] = v > 0.f ? v : 0.f;
    }
  }
}

// ---------------------------------------------------------------------------
extern "C" void kernel_launch(void* const* d_in, const int* in_sizes, int n_in,
                              void* d_out, int out_size, void* d_ws, size_t ws_size,
                              hipStream_t stream) {
  (void)in_sizes; (void)n_in; (void)out_size; (void)ws_size;
  // setup_inputs order:
  // 0 nodes[int N] 1 features[f32 N*128] 2 neigh_idx[int 2*N*10]
  // 3 neigh_mask[f32 2*N*10] 4 W1[f32 256*256] 5 b1[f32 256]
  // 6 W2[f32 512*256] 7 b2[f32 256]
  const float* features   = (const float*)d_in[1];
  const int*   neigh_idx  = (const int*)d_in[2];
  const float* neigh_mask = (const float*)d_in[3];
  const float* W1 = (const float*)d_in[4];
  const float* b1 = (const float*)d_in[5];
  const float* W2 = (const float*)d_in[6];
  const float* b2 = (const float*)d_in[7];
  float* out = (float*)d_out;

  // Workspace layout: h [N,256] f32 | Wt1 bf16 [256,256] | Wt2 bf16 [256,512]
  char* ws = (char*)d_ws;
  const size_t hbytes = (size_t)NNODES * DOUT * sizeof(float);
  float*  h   = (float*)ws;
  bf16_t* Wt1 = (bf16_t*)(ws + hbytes);
  bf16_t* Wt2 = (bf16_t*)(ws + hbytes + (size_t)256 * 256 * sizeof(bf16_t));

  wt_convert_kernel<<<(256 * 256 + 255) / 256, 256, 0, stream>>>(W1, Wt1, 256, 256);
  wt_convert_kernel<<<(512 * 256 + 255) / 256, 256, 0, stream>>>(W2, Wt2, 512, 256);

  sage_layer_kernel<128><<<NNODES / 16, 256, 0, stream>>>(
      features, neigh_idx, neigh_mask, Wt1, b1, h);
  sage_layer_kernel<256><<<NNODES / 16, 256, 0, stream>>>(
      h, neigh_idx + (size_t)NNODES * SS, neigh_mask + (size_t)NNODES * SS,
      Wt2, b2, out);
}
